// HeteroGCN_80470507258386
// MI455X (gfx1250) — matrix operand order repaired and checked by reading, not compiled, not run on hardware
//
#include <hip/hip_runtime.h>
#include <hip/hip_bf16.h>
#include <math.h>

#define HDIM 128
#define NNODE 65536
#define NZERO 32768
#define RREL 4
#define NEDGE 524288
#define BSZ 1024
#define PER 64
#define NLAYER 2

typedef __attribute__((ext_vector_type(16))) __bf16 v16bf;
typedef __attribute__((ext_vector_type(8)))  __bf16 v8bf;
typedef __attribute__((ext_vector_type(4)))  __bf16 v4bf;
typedef __attribute__((ext_vector_type(8)))  float  v8f;
typedef __attribute__((ext_vector_type(4)))  unsigned int u32x4;
typedef __attribute__((ext_vector_type(8)))  int  i32x8;
typedef __attribute__((ext_vector_type(4)))  int  i32x4;

#if defined(__HIP_DEVICE_COMPILE__) && __has_builtin(__builtin_amdgcn_tensor_load_to_lds)
#define HAVE_TDM 1
#else
#define HAVE_TDM 0
#endif

#if __has_include(<hip/amd_detail/amd_gfx1250_TDM.h>)
#define TDM_ARITY6 1
#else
#define TDM_ARITY6 0
#endif

// ---------------------------------------------------------------------------
// TDM: DMA one contiguous 4 KB tile (1024 dwords) from global to LDS.
// D# built per CDNA5 ISA 8.3/8.4: group0 {count=1, lds_addr, global_addr,
// type=2}; group1 {data_size=4B, tensor_dim0=tile_dim0=1024, tensor_dim1=1,
// tile_dim1=1, dim0_stride=1024}.
// ---------------------------------------------------------------------------
#if HAVE_TDM
__device__ __forceinline__ void tdm_load_tile4k(const void* gsrc, unsigned lds_off)
{
    unsigned long long ga = (unsigned long long)gsrc;
    u32x4 g0;
    g0[0] = 1u;                                              // count=1
    g0[1] = lds_off;                                         // lds_addr (bytes)
    g0[2] = (unsigned)ga;                                    // global_addr[31:0]
    g0[3] = ((unsigned)(ga >> 32) & 0x01ffffffu) | 0x80000000u; // addr[56:32] | type=2
    i32x8 g1;
    g1[0] = 0x00020000;      // data_size = 4B
    g1[1] = 1024 << 16;      // tensor_dim0[15:0]
    g1[2] = 1 << 16;         // tensor_dim0[31:16]=0 | tensor_dim1[15:0]=1
    g1[3] = 1024 << 16;      // tensor_dim1[31:16]=0 | tile_dim0=1024
    g1[4] = 1;               // tile_dim1=1 | tile_dim2=0
    g1[5] = 1024;            // tensor_dim0_stride[31:0]
    g1[6] = 0;               // stride hi | dim1_stride lo
    g1[7] = 0;
    i32x4 g2 = {0, 0, 0, 0};
    i32x4 g3 = {0, 0, 0, 0};
#if TDM_ARITY6
    i32x8 g4 = {0, 0, 0, 0, 0, 0, 0, 0};
    __builtin_amdgcn_tensor_load_to_lds(g0, g1, g2, g3, g4, 0);
#else
    __builtin_amdgcn_tensor_load_to_lds(g0, g1, g2, g3, 0);
#endif
}
#endif

// ---------------------------------------------------------------------------
// Weight packing: fp32 [K=H][N=H] row-major -> WMMA-B bf16 layout.
// B 32x16 tile: lane l holds col n = nt*16 + (l&15), k = 32*kt + (l>>4)*16 + j.
// ---------------------------------------------------------------------------
__global__ void pack_w_kernel(const float* __restrict__ W0, const float* __restrict__ W1,
                              const float* __restrict__ fcW, const float* __restrict__ conv_root,
                              const float* __restrict__ conv_rel, const float* __restrict__ attW,
                              __bf16* __restrict__ Wp)
{
    int y   = blockIdx.y;                                   // 0..20 matrix id
    int idx = blockIdx.x * blockDim.x + threadIdx.x;        // 0..16383
    const float* src;
    if      (y == 0) src = W0;
    else if (y == 1) src = W1;
    else if (y == 2) src = fcW;
    else {
        int t = y - 3, l = t / 9, u = t % 9;
        if      (u < 4) src = conv_rel  + (long)(l * RREL + u)     * HDIM * HDIM;
        else if (u < 8) src = attW      + (long)(l * 16 + (u - 4)) * HDIM * HDIM;
        else            src = conv_root + (long)l                  * HDIM * HDIM;
    }
    int k  = idx >> 7, n = idx & 127;
    int kt = k >> 5,  kl = k & 31;
    int g  = kl >> 4, j  = kl & 15;
    int lane = g * 16 + (n & 15), nt = n >> 4;
    Wp[(long)y * HDIM * HDIM + (((kt * 8 + nt) * 32 + lane) * 16 + j)] = (__bf16)src[idx];
}

// ---------------------------------------------------------------------------
// Activation packing: fp32 [N][H] -> WMMA-A bf16 layout, 8 elements/thread.
// A 16x32 tile: lane l holds row m = l&15, k = 32*kt + 16*(l>>4) + j.
// Contiguous 8-element runs: chunk c: kt=c>>2, g=(c>>1)&1, half=c&1,
// k = 32*kt + 16*g + 8*half + i  <->  j = half*8 + i.
// mode 0: concat(x0,x1); mode 1: s0*mask; mode 2: s0+s1
// ---------------------------------------------------------------------------
__global__ void pack_a_kernel(const float* __restrict__ s0, const float* __restrict__ s1,
                              const float* __restrict__ mask, __bf16* __restrict__ Ap, int mode)
{
    int t = blockIdx.x * blockDim.x + threadIdx.x;   // N*16 threads
    int n = t >> 4, c = t & 15;
    int kt = c >> 2, g = (c >> 1) & 1, half = c & 1;
    int kbase = 32 * kt + 16 * g + 8 * half;
    long soff = (long)n * HDIM + kbase;
    float v[8];
    if (mode == 0) {
        const float* s = (n < NZERO) ? (s0 + soff) : (s1 + (soff - (long)NZERO * HDIM));
#pragma unroll
        for (int i = 0; i < 8; ++i) v[i] = s[i];
    } else if (mode == 1) {
#pragma unroll
        for (int i = 0; i < 8; ++i) v[i] = s0[soff + i] * mask[soff + i];
    } else {
#pragma unroll
        for (int i = 0; i < 8; ++i) v[i] = s0[soff + i] + s1[soff + i];
    }
    int lane = g * 16 + (n & 15);
    v8bf o;
#pragma unroll
    for (int i = 0; i < 8; ++i) o[i] = (__bf16)v[i];
    *(v8bf*)(Ap + (((long)(n >> 4) * 4 + kt) * 32 + lane) * 16 + half * 8) = o;
}

// ---------------------------------------------------------------------------
// WMMA GEMM: C[16 x 128] per block (8 waves, one 16x16 tile each), K=128 in
// four 16x16x32 bf16 WMMA steps. A tile (4 KB) staged in LDS via TDM by wave 0
// (s_wait_tensorcnt + barrier), consumed via ds_load; B streamed from global.
// mode 0: fp32 out; mode 1: bf16 out; mode 2: fp32 out with final permutation.
// ---------------------------------------------------------------------------
__global__ void wmma_gemm_kernel(const __bf16* __restrict__ Ap, const __bf16* __restrict__ Wp,
                                 void* __restrict__ outBase, const float* __restrict__ bias,
                                 int mode, long outZstride)
{
    __shared__ __align__(32) __bf16 sA[16 * HDIM];   // 4 KB A tile
    int mt   = blockIdx.x;
    int z    = blockIdx.z;
    int wv   = threadIdx.x >> 5;     // 0..7 -> column tile
    int lane = threadIdx.x & 31;
    const __bf16* atile = Ap + (long)mt * (16 * HDIM);
#if HAVE_TDM
    if (threadIdx.x < 32) {
        tdm_load_tile4k(atile, (unsigned)(unsigned long long)(size_t)sA);
        __builtin_amdgcn_s_wait_tensorcnt(0);
    }
    __syncthreads();
#else
    ((uint4*)sA)[threadIdx.x] = ((const uint4*)atile)[threadIdx.x];  // 256*16B = 4KB
    __syncthreads();
#endif
    const __bf16* W = Wp + (long)z * HDIM * HDIM;
    v8f acc = {};
#pragma unroll
    for (int kt = 0; kt < 4; ++kt) {
        v16bf a = *(const v16bf*)(sA + (kt * 32 + lane) * 16);
        v16bf b = *(const v16bf*)(W + (((kt * 8 + wv) * 32 + lane) * 16));
        acc = __builtin_amdgcn_wmma_f32_16x16x32_bf16(false, a, false, b,
                                                      (short)0, acc, false, false);
    }
    int col   = wv * 16 + (lane & 15);
    int rbase = mt * 16 + ((lane >> 4) << 3);
    float bv  = (bias != nullptr) ? bias[col] : 0.0f;
    if (mode == 0) {
        float* out = (float*)outBase + z * outZstride;
#pragma unroll
        for (int i = 0; i < 8; ++i) out[(long)(rbase + i) * HDIM + col] = acc[i] + bv;
    } else if (mode == 1) {
        __bf16* out = (__bf16*)outBase + z * outZstride;
#pragma unroll
        for (int i = 0; i < 8; ++i) out[(long)(rbase + i) * HDIM + col] = (__bf16)(acc[i] + bv);
    } else {
        float* out = (float*)outBase;
#pragma unroll
        for (int i = 0; i < 8; ++i) {
            int n = rbase + i;
            int bb = n >> 6, p = n & 63;
            long off = (p == 0) ? (long)bb * HDIM
                                : (long)BSZ * HDIM + ((long)bb * (PER - 1) + (p - 1)) * HDIM;
            out[off + col] = acc[i] + bv;
        }
    }
}

// ---------------------------------------------------------------------------
// Attention scalars, wave-per-(r,n): sq[row] = <xa[row,:], att_q[r,:]>.
// Lane handles 4 channels (b64 bf16 gather + float4 weights), shfl reduce.
// ---------------------------------------------------------------------------
__global__ void sqsk_kernel(const __bf16* __restrict__ xa, const float* __restrict__ attq,
                            const float* __restrict__ attk, float* __restrict__ sq,
                            float* __restrict__ sk)
{
    int t    = blockIdx.x * blockDim.x + threadIdx.x;
    int row  = t >> 5;               // r*NNODE + n
    int lane = t & 31;
    int hb   = lane * 4;
    int r    = row >> 16;            // NNODE == 65536
    v4bf v = *(const v4bf*)(xa + (long)row * HDIM + hb);
    float4 q = *(const float4*)(attq + r * HDIM + hb);
    float4 k = *(const float4*)(attk + r * HDIM + hb);
    float a = (float)v[0] * q.x + (float)v[1] * q.y + (float)v[2] * q.z + (float)v[3] * q.w;
    float b = (float)v[0] * k.x + (float)v[1] * k.y + (float)v[2] * k.z + (float)v[3] * k.w;
    for (int off = 16; off > 0; off >>= 1) {
        a += __shfl_down(a, off, 32);
        b += __shfl_down(b, off, 32);
    }
    if (lane == 0) { sq[row] = a; sk[row] = b; }
}

// ---------------------------------------------------------------------------
// Per-layer init kernels
// ---------------------------------------------------------------------------
__global__ void init_seg_kernel(int* __restrict__ cnt, float* __restrict__ mx,
                                float* __restrict__ den)
{
    int i = blockIdx.x * blockDim.x + threadIdx.x;   // N*R threads
    cnt[i] = 0;
    mx[i]  = -__builtin_inff();
    den[i] = 0.0f;
}

__global__ void init_att_kernel(const float* __restrict__ attb, float* __restrict__ xatt)
{
    long i = (long)blockIdx.x * blockDim.x + threadIdx.x;  // N*H threads
    xatt[i] = attb[(int)(i & 127)];
}

__global__ void init_bn_kernel(float* __restrict__ s, float* __restrict__ q)
{
    int i = threadIdx.x;  // 128
    s[i] = 0.0f; q[i] = 0.0f;
}

// ---------------------------------------------------------------------------
// Edge kernels.  conv/att message kernels: wave-per-edge, 4 channels/lane.
// ---------------------------------------------------------------------------
__global__ void count_edges_kernel(const int* __restrict__ dst, const int* __restrict__ et,
                                   int* __restrict__ cnt)
{
    int e = blockIdx.x * blockDim.x + threadIdx.x;
    atomicAdd(&cnt[dst[e] * RREL + et[e]], 1);
}

__global__ void conv_msg_kernel(const __bf16* __restrict__ xr, const int* __restrict__ src,
                                const int* __restrict__ dst, const int* __restrict__ et,
                                const int* __restrict__ cnt, float* __restrict__ xconv)
{
    int t = blockIdx.x * blockDim.x + threadIdx.x;
    int e = t >> 5, hb = (t & 31) * 4;
    int s = src[e], d = dst[e], r = et[e];
    int c = cnt[d * RREL + r];
    float inv = 1.0f / (float)(c > 0 ? c : 1);
    v4bf v = *(const v4bf*)(xr + ((long)r * NNODE + s) * HDIM + hb);
    float* o = xconv + (long)d * HDIM + hb;
    atomicAdd(o + 0, (float)v[0] * inv);
    atomicAdd(o + 1, (float)v[1] * inv);
    atomicAdd(o + 2, (float)v[2] * inv);
    atomicAdd(o + 3, (float)v[3] * inv);
}

__device__ __forceinline__ void atomicMaxF(float* a, float v)
{
    unsigned int* ai = (unsigned int*)a;
    float cur = __uint_as_float(*ai);
    while (v > cur) {
        unsigned int assumed = __float_as_uint(cur);
        unsigned int prev = atomicCAS(ai, assumed, __float_as_uint(v));
        if (prev == assumed) break;
        cur = __uint_as_float(prev);
    }
}

__global__ void logit_max_kernel(const float* __restrict__ sq, const float* __restrict__ sk,
                                 const int* __restrict__ src, const int* __restrict__ dst,
                                 const int* __restrict__ et, float* __restrict__ logit,
                                 float* __restrict__ mx)
{
    int e = blockIdx.x * blockDim.x + threadIdx.x;
    int r = et[e], d = dst[e], s = src[e];
    float v = sq[(long)r * NNODE + d] + sk[(long)r * NNODE + s];
    v = (v > 0.0f) ? v : 0.2f * v;             // leaky_relu(0.2)
    logit[e] = v;
    atomicMaxF(&mx[d * RREL + r], v);
}

__global__ void exp_den_kernel(const float* __restrict__ logit, const float* __restrict__ mx,
                               const int* __restrict__ dst, const int* __restrict__ et,
                               float* __restrict__ exb, float* __restrict__ den)
{
    int e = blockIdx.x * blockDim.x + threadIdx.x;
    int seg = dst[e] * RREL + et[e];
    float ex = expf(logit[e] - mx[seg]);
    exb[e] = ex;
    atomicAdd(&den[seg], ex);
}

__global__ void att_msg_kernel(const __bf16* __restrict__ xa, const int* __restrict__ src,
                               const int* __restrict__ dst, const int* __restrict__ et,
                               const float* __restrict__ exb, const float* __restrict__ den,
                               float* __restrict__ xatt)
{
    int t = blockIdx.x * blockDim.x + threadIdx.x;
    int e = t >> 5, hb = (t & 31) * 4;
    int s = src[e], d = dst[e], r = et[e];
    float alpha = exb[e] / fmaxf(den[d * RREL + r], 1e-16f);
    v4bf v = *(const v4bf*)(xa + ((long)r * NNODE + s) * HDIM + hb);
    float* o = xatt + (long)d * HDIM + hb;
    atomicAdd(o + 0, alpha * (float)v[0]);
    atomicAdd(o + 1, alpha * (float)v[1]);
    atomicAdd(o + 2, alpha * (float)v[2]);
    atomicAdd(o + 3, alpha * (float)v[3]);
}

// ---------------------------------------------------------------------------
// BatchNorm (training stats) + leaky_relu(0.01)
// ---------------------------------------------------------------------------
__global__ void bn_stats_kernel(const float* __restrict__ xc, const float* __restrict__ xa,
                                float* __restrict__ bsum, float* __restrict__ bsq)
{
    int h = threadIdx.x;  // 128
    float s = 0.0f, q = 0.0f;
    for (int n = blockIdx.x; n < NNODE; n += gridDim.x) {
        float v = xc[(long)n * HDIM + h] + xa[(long)n * HDIM + h];
        s += v; q += v * v;
    }
    atomicAdd(&bsum[h], s);
    atomicAdd(&bsq[h], q);
}

__global__ void bn_apply_kernel(const float* __restrict__ xc, const float* __restrict__ xa,
                                const float* __restrict__ bsum, const float* __restrict__ bsq,
                                const float* __restrict__ g, const float* __restrict__ b,
                                float* __restrict__ x)
{
    int n = blockIdx.x, h = threadIdx.x;
    const float invN = 1.0f / (float)NNODE;
    float mu  = bsum[h] * invN;
    float var = bsq[h] * invN - mu * mu;
    float v = xc[(long)n * HDIM + h] + xa[(long)n * HDIM + h];
    float y = (v - mu) * rsqrtf(var + 1e-5f) * g[h] + b[h];
    x[(long)n * HDIM + h] = (y > 0.0f) ? y : 0.01f * y;
}

// ---------------------------------------------------------------------------
// Host launch
// ---------------------------------------------------------------------------
extern "C" void kernel_launch(void* const* d_in, const int* in_sizes, int n_in,
                              void* d_out, int out_size, void* d_ws, size_t ws_size,
                              hipStream_t stream)
{
    const float* x0        = (const float*)d_in[0];
    const float* x1        = (const float*)d_in[1];
    const float* W0        = (const float*)d_in[2];
    const float* b0        = (const float*)d_in[3];
    const float* W1        = (const float*)d_in[4];
    const float* b1        = (const float*)d_in[5];
    const float* mask      = (const float*)d_in[6];
    const float* conv_root = (const float*)d_in[7];
    const float* conv_rel  = (const float*)d_in[8];
    const float* conv_b    = (const float*)d_in[9];
    const float* attW      = (const float*)d_in[10];
    const float* att_q     = (const float*)d_in[11];
    const float* att_k     = (const float*)d_in[12];
    const float* att_b     = (const float*)d_in[13];
    const float* bn_g      = (const float*)d_in[14];
    const float* bn_b      = (const float*)d_in[15];
    const float* fc_W      = (const float*)d_in[16];
    const float* fc_b      = (const float*)d_in[17];
    const int*   eidx      = (const int*)d_in[18];
    const int*   etype     = (const int*)d_in[19];
    const int*   srcv      = eidx;
    const int*   dstv      = eidx + NEDGE;

    // workspace carve-out (256B aligned)
    char* w = (char*)d_ws;
    size_t off = 0;
    auto carve = [&](size_t bytes) -> void* {
        off = (off + 255) & ~(size_t)255;
        void* p = w + off;
        off += bytes;
        return p;
    };
    __bf16* Ap    = (__bf16*)carve((size_t)NNODE * HDIM * 2);          // 16 MB
    __bf16* Wp    = (__bf16*)carve((size_t)21 * HDIM * HDIM * 2);      // 672 KB
    float*  x_in  = (float*) carve((size_t)NNODE * HDIM * 4);          // 32 MB
    float*  x     = (float*) carve((size_t)NNODE * HDIM * 4);          // 32 MB
    float*  xconv = (float*) carve((size_t)NNODE * HDIM * 4);          // 32 MB
    float*  xatt  = (float*) carve((size_t)NNODE * HDIM * 4);          // 32 MB
    __bf16* xrxa  = (__bf16*)carve((size_t)8 * NNODE * HDIM * 2);      // 128 MB (xr[0..3], xa[0..3])
    float*  sq    = (float*) carve((size_t)RREL * NNODE * 4);
    float*  sk    = (float*) carve((size_t)RREL * NNODE * 4);
    int*    cnt   = (int*)   carve((size_t)NNODE * RREL * 4);
    float*  mx    = (float*) carve((size_t)NNODE * RREL * 4);
    float*  den   = (float*) carve((size_t)NNODE * RREL * 4);
    float*  logit = (float*) carve((size_t)NEDGE * 4);
    float*  exb   = (float*) carve((size_t)NEDGE * 4);
    float*  bsum  = (float*) carve((size_t)HDIM * 4);
    float*  bsq   = (float*) carve((size_t)HDIM * 4);

    const long NH = (long)NNODE * HDIM;

    // weights -> WMMA-B bf16 layout (21 matrices)
    pack_w_kernel<<<dim3(64, 21), 256, 0, stream>>>(W0, W1, fc_W, conv_root, conv_rel, attW, Wp);

    // x_in = [x0@W0+b0 ; x1@W1+b1]
    pack_a_kernel<<<4096, 256, 0, stream>>>(x0, x1, nullptr, Ap, 0);
    wmma_gemm_kernel<<<dim3(2048, 1, 1), 256, 0, stream>>>(Ap, Wp, x_in, b0, 0, 0);
    wmma_gemm_kernel<<<dim3(2048, 1, 1), 256, 0, stream>>>(Ap + (long)NZERO * HDIM,
                                                           Wp + (long)1 * HDIM * HDIM,
                                                           x_in + (long)NZERO * HDIM, b1, 0, 0);

    const float* xprev = x_in;
    for (int l = 0; l < NLAYER; ++l) {
        const __bf16* wl = Wp + (long)(3 + l * 9) * HDIM * HDIM;
        // xm = x*mask -> packed A
        pack_a_kernel<<<4096, 256, 0, stream>>>(xprev, nullptr, mask, Ap, 1);
        // xr[0..3], xa[0..3] = xm @ {conv_rel, attW}   (bf16 out)
        wmma_gemm_kernel<<<dim3(4096, 1, 8), 256, 0, stream>>>(Ap, wl, xrxa, nullptr, 1, NH);
        // x_conv = xm @ conv_root + conv_b
        wmma_gemm_kernel<<<dim3(4096, 1, 1), 256, 0, stream>>>(Ap, wl + (long)8 * HDIM * HDIM,
                                                               xconv, conv_b + l * HDIM, 0, 0);
        // attention scalars
        sqsk_kernel<<<32768, 256, 0, stream>>>(xrxa + 4 * NH,
                                               att_q + (long)l * 16 * HDIM,
                                               att_k + (long)l * 16 * HDIM, sq, sk);
        // init per-layer state
        init_seg_kernel<<<(NNODE * RREL) / 256, 256, 0, stream>>>(cnt, mx, den);
        init_att_kernel<<<32768, 256, 0, stream>>>(att_b + l * HDIM, xatt);
        init_bn_kernel<<<1, 128, 0, stream>>>(bsum, bsq);
        // RGCN mean aggregation: per-edge m/cnt[seg] folded into x_conv
        count_edges_kernel<<<NEDGE / 256, 256, 0, stream>>>(dstv, etype, cnt);
        conv_msg_kernel<<<NEDGE / 8, 256, 0, stream>>>(xrxa, srcv, dstv, etype, cnt, xconv);
        // RGAT within-relation softmax
        logit_max_kernel<<<NEDGE / 256, 256, 0, stream>>>(sq, sk, srcv, dstv, etype, logit, mx);
        exp_den_kernel<<<NEDGE / 256, 256, 0, stream>>>(logit, mx, dstv, etype, exb, den);
        att_msg_kernel<<<NEDGE / 8, 256, 0, stream>>>(xrxa + 4 * NH, srcv, dstv, etype,
                                                      exb, den, xatt);
        // BN + leaky_relu(0.01)
        bn_stats_kernel<<<512, 128, 0, stream>>>(xconv, xatt, bsum, bsq);
        bn_apply_kernel<<<NNODE, 128, 0, stream>>>(xconv, xatt, bsum, bsq,
                                                   bn_g + l * HDIM, bn_b + l * HDIM, x);
        xprev = x;
    }

    // out = (x + x_in) @ fc_W + fc_b, written permuted into d_out
    pack_a_kernel<<<4096, 256, 0, stream>>>(x, x_in, nullptr, Ap, 2);
    wmma_gemm_kernel<<<dim3(4096, 1, 1), 256, 0, stream>>>(Ap, Wp + (long)2 * HDIM * HDIM,
                                                           d_out, fc_b, 2, 0);
    (void)in_sizes; (void)n_in; (void)out_size; (void)ws_size;
}
